// STU_13134009991271
// MI455X (gfx1250) — matrix operand order repaired and checked by reading, not compiled, not run on hardware
//
#include <hip/hip_runtime.h>

// ---------------- types ----------------
typedef __bf16 bf16_t;
typedef __attribute__((ext_vector_type(16))) __bf16    v16bf;
typedef __attribute__((ext_vector_type(8)))  float     v8f;
typedef __attribute__((ext_vector_type(8)))  unsigned  v8u;
typedef __attribute__((ext_vector_type(4)))  unsigned  v4u;
typedef __attribute__((ext_vector_type(8)))  int       v8i;
typedef __attribute__((ext_vector_type(4)))  int       v4i;

#if __has_builtin(__builtin_amdgcn_tensor_load_to_lds)
#define HAVE_TDM 1
#else
#define HAVE_TDM 0
#endif

static __device__ __forceinline__ bf16_t f2bf(float f) {
  unsigned u = __builtin_bit_cast(unsigned, f);
  unsigned r = (u + 0x7FFFu + ((u >> 16) & 1u)) >> 16;   // round-to-nearest-even
  return __builtin_bit_cast(bf16_t, (unsigned short)r);
}
static __device__ __forceinline__ unsigned bf_bits(bf16_t v) {
  return (unsigned)__builtin_bit_cast(unsigned short, v);
}

#if HAVE_TDM
// LDS byte offset of a __shared__ object (addrspacecast generic->LDS, then ptrtoint).
static __device__ __forceinline__ unsigned lds_offset(const void* p) {
  return (unsigned)(unsigned long long)(__attribute__((address_space(3))) const void*)p;
}

// Issue a TDM 2-D tile load: tile_d0 elements (contiguous, data_size=2B) x tile_d1 rows,
// row stride = stride0 elements, packed row-major into LDS at lds_addr.
static __device__ __forceinline__ void tdm_load_2d_bf16(
    const void* gaddr, unsigned lds_addr,
    unsigned tensor_d0, unsigned tensor_d1,
    unsigned tile_d0, unsigned tile_d1,
    unsigned long long stride0_elems)
{
  unsigned long long ga = (unsigned long long)(size_t)gaddr;
  v4u g0;
  g0[0] = 1u;                                             // count=1, user descriptor
  g0[1] = lds_addr;                                       // lds_addr[31:0]
  g0[2] = (unsigned)(ga & 0xFFFFFFFFu);                   // global_addr[31:0]
  g0[3] = (unsigned)((ga >> 32) & 0x1FFFFFFu) | (2u << 30); // addr[56:32] | type=2
  v8i g1;
  g1[0] = (int)(1u << 16);                                // workgroup_mask=0, data_size=1 (2B)
  g1[1] = (int)((tensor_d0 & 0xFFFFu) << 16);             // tensor_dim0[15:0]
  g1[2] = (int)(((tensor_d0 >> 16) & 0xFFFFu) | ((tensor_d1 & 0xFFFFu) << 16));
  g1[3] = (int)(((tensor_d1 >> 16) & 0xFFFFu) | ((tile_d0 & 0xFFFFu) << 16));
  g1[4] = (int)(tile_d1 & 0xFFFFu);                       // tile_dim1, tile_dim2=0
  g1[5] = (int)(stride0_elems & 0xFFFFFFFFull);           // tensor_dim0_stride[31:0]
  g1[6] = (int)((stride0_elems >> 32) & 0xFFFFull);       // stride[47:32]
  g1[7] = 0;
  v4i z4 = {0, 0, 0, 0};
  v8i z8 = {0, 0, 0, 0, 0, 0, 0, 0};
  __builtin_amdgcn_tensor_load_to_lds(g0, g1, z4, z4, z8, 0);
}

static __device__ __forceinline__ void tdm_wait0() {
#if __has_builtin(__builtin_amdgcn_s_wait_tensorcnt)
  __builtin_amdgcn_s_wait_tensorcnt(0);
#else
  asm volatile("s_wait_tensorcnt 0x0" ::: "memory");
#endif
}
#endif

// ---------------- tiling ----------------
#define TM 128
#define TN 128
#define KSTEP 32
#define SA   40  // conv A-tile LDS stride (elements)
#define SA2  32  // gemm2 A-tile LDS stride (TDM packs rows contiguously)
#define SBT  40  // transposed B-tile LDS stride (elements)

// A fragment (16x32 bf16) from LDS row-major tile.
static __device__ __forceinline__ v16bf ldsA16x32(const bf16_t* base, int stride, int lane) {
  const int row  = lane & 15;
  const int half = lane >> 4;
  const bf16_t* p = base + row * stride + half * 8;
  uint4 a = *reinterpret_cast<const uint4*>(p);
  uint4 b = *reinterpret_cast<const uint4*>(p + 16);
  v8u r;
  r[0]=a.x; r[1]=a.y; r[2]=a.z; r[3]=a.w;
  r[4]=b.x; r[5]=b.y; r[6]=b.z; r[7]=b.w;
  return __builtin_bit_cast(v16bf, r);
}

// B fragment (32x16 bf16) from LDS transposed tile BT[col][row].
static __device__ __forceinline__ v16bf ldsB32x16(const bf16_t* baseT, int strideT, int lane) {
  const int col  = lane & 15;
  const int half = lane >> 4;
  const bf16_t* p = baseT + col * strideT + half * 8;
  uint4 a = *reinterpret_cast<const uint4*>(p);
  uint4 b = *reinterpret_cast<const uint4*>(p + 16);
  v8u r;
  r[0]=a.x; r[1]=a.y; r[2]=a.z; r[3]=a.w;
  r[4]=b.x; r[5]=b.y; r[6]=b.z; r[7]=b.w;
  return __builtin_bit_cast(v16bf, r);
}

// ---------------- prep kernels ----------------
__global__ void stu_zero_out(float* __restrict__ p, int n) {
  int i = blockIdx.x * blockDim.x + threadIdx.x;
  if (i < n) p[i] = 0.0f;
}

__global__ void stu_prep_x(const float* __restrict__ x, bf16_t* __restrict__ xb,
                           bf16_t* __restrict__ U, int L, int D, int KW, int kuOff) {
  int idx = blockIdx.x * blockDim.x + threadIdx.x;
  if (idx >= L * D) return;
  int l = idx / D, d = idx - l * D;
  xb[idx] = f2bf(x[idx]);
#pragma unroll
  for (int i = 0; i < 3; ++i) {
    float s = (l - i >= 0) ? x[(l - i) * D + d] : 0.0f;
    U[(size_t)l * KW + kuOff + i * D + d] = f2bf(s);
  }
}

__global__ void stu_prep_phi(const float* __restrict__ phi, bf16_t* __restrict__ phis,
                             int L, int K) {
  int idx = blockIdx.x * blockDim.x + threadIdx.x;
  if (idx >= 2 * K * L) return;
  int sk = idx / L, j = idx - sk * L;
  int s = sk / K, k = sk - s * K;
  float v = phi[j * K + k];
  if (s == 1 && (j & 1)) v = -v;
  phis[sk * L + j] = f2bf(v);
}

__global__ void stu_prep_w(const float* __restrict__ Mp, const float* __restrict__ Mm,
                           const float* __restrict__ Mu, const float* __restrict__ sigma,
                           bf16_t* __restrict__ W, int K, int D, int KW) {
  int idx = blockIdx.x * blockDim.x + threadIdx.x;
  if (idx >= KW * D) return;
  int r = idx / D, o = idx - r * D;
  float v;
  int kd2 = 2 * K * D;
  if (r < kd2) {
    int s = r / (K * D);
    int rem = r - s * (K * D);
    int k = rem / D, d = rem - k * D;
    float sig4 = sqrtf(sqrtf(sigma[k]));
    const float* M = s ? Mm : Mp;
    v = M[((size_t)k * D + d) * D + o] * sig4;
  } else {
    int rem = r - kd2;
    int i = rem / D, d = rem - i * D;
    v = Mu[((size_t)i * D + d) * D + o];
  }
  W[(size_t)r * D + o] = f2bf(v);
}

// ---------------- stage 2: causal Toeplitz conv as WMMA GEMM (double-buffered) --
// U[l, sk*D + d] = sum_{j<=l} phis[sk][l-j] * xb[j, d]
__global__ __launch_bounds__(256) void stu_conv_kernel(
    const bf16_t* __restrict__ xb,    // (L, D)
    const bf16_t* __restrict__ phis,  // (2K, L)
    bf16_t* __restrict__ U,           // (L, KW)
    int L, int D, int KW) {
  __shared__ alignas(16) bf16_t ph[2048];
  __shared__ alignas(16) bf16_t As[2][TM * SA];
  __shared__ alignas(16) bf16_t Bt[2][TN * SBT];

  const int n0 = blockIdx.x * TN;
  const int l0 = blockIdx.y * TM;
  const int sk = blockIdx.z;
  const bf16_t* __restrict__ phir = phis + sk * L;

  const int tid  = threadIdx.x;
  const int lane = tid & 31;
  const int wave = tid >> 5;
  const int wm = wave & 3;
  const int wn = wave >> 2;

  // stage filter row into LDS once
  for (int c = tid; c < L / 8; c += 256)
    *reinterpret_cast<uint4*>(&ph[c * 8]) =
        *reinterpret_cast<const uint4*>(phir + c * 8);
  __syncthreads();

  // Toeplitz A-tile generator (branch-free, packed b128 stores)
  auto stageA = [&](int j0, int buf) {
    const int base = l0 - j0;
    for (int e = tid; e < (TM * KSTEP) / 8; e += 256) {
      int i = e >> 2, jg = e & 3;
      int d0 = base + i - jg * 8;
      unsigned w[4];
#pragma unroll
      for (int t = 0; t < 4; ++t) {
        int da = d0 - 2 * t, db = da - 1;
        unsigned lo = bf_bits(ph[da < 0 ? 0 : da]);
        unsigned hi = bf_bits(ph[db < 0 ? 0 : db]);
        lo = (da >= 0) ? lo : 0u;
        hi = (db >= 0) ? hi : 0u;
        w[t] = lo | (hi << 16);
      }
      uint4 vv; vv.x = w[0]; vv.y = w[1]; vv.z = w[2]; vv.w = w[3];
      *reinterpret_cast<uint4*>(&As[buf][i * SA + jg * 8]) = vv;
    }
  };
  // transposed x-tile: Bt[c][r] = xb[j0+r][n0+c]
  auto stageB = [&](int j0, int buf) {
    for (int e = tid; e < (KSTEP * TN) / 8; e += 256) {
      int r = e >> 4, cg = e & 15;
      uint4 v = *reinterpret_cast<const uint4*>(xb + (j0 + r) * D + n0 + cg * 8);
      unsigned w[4] = {v.x, v.y, v.z, v.w};
#pragma unroll
      for (int t = 0; t < 8; ++t) {
        unsigned short h = (unsigned short)((w[t >> 1] >> ((t & 1) * 16)) & 0xFFFFu);
        Bt[buf][(cg * 8 + t) * SBT + r] = __builtin_bit_cast(bf16_t, h);
      }
    }
  };

  v8f acc[2][4];
#pragma unroll
  for (int i = 0; i < 2; ++i)
#pragma unroll
    for (int j = 0; j < 4; ++j) acc[i][j] = (v8f){0,0,0,0,0,0,0,0};

  const int jend  = min(L, l0 + TM);     // causality
  const int niter = jend / KSTEP;

  stageA(0, 0);
  stageB(0, 0);
  __syncthreads();

  for (int it = 0; it < niter; ++it) {
    const int buf = it & 1;
    if (it + 1 < niter) {                // prepare next tiles while computing
      int jn = (it + 1) * KSTEP;
      stageA(jn, buf ^ 1);
      stageB(jn, buf ^ 1);
      __builtin_prefetch(xb + (jn + KSTEP < jend ? jn + KSTEP : jn) * D + n0, 0, 1);
    }
    v16bf afr[2];
    afr[0] = ldsA16x32(&As[buf][(wm * 32 + 0)  * SA], SA, lane);
    afr[1] = ldsA16x32(&As[buf][(wm * 32 + 16) * SA], SA, lane);
#pragma unroll
    for (int fn = 0; fn < 4; ++fn) {
      v16bf bfr = ldsB32x16(&Bt[buf][(wn * 64 + fn * 16) * SBT], SBT, lane);
#pragma unroll
      for (int fm = 0; fm < 2; ++fm)
        acc[fm][fn] = __builtin_amdgcn_wmma_f32_16x16x32_bf16(
            false, afr[fm], false, bfr, (short)0, acc[fm][fn], false, false);
    }
    if (it + 1 < niter) __syncthreads();
  }

  const int half = lane >> 4, lcol = lane & 15;
#pragma unroll
  for (int fm = 0; fm < 2; ++fm)
#pragma unroll
    for (int fn = 0; fn < 4; ++fn) {
      int r0 = l0 + wm * 32 + fm * 16 + half * 8;
      int c  = sk * D + n0 + wn * 64 + fn * 16 + lcol;
#pragma unroll
      for (int r = 0; r < 8; ++r)
        U[(size_t)(r0 + r) * KW + c] = f2bf(acc[fm][fn][r]);
    }
}

// -------- stage 3: out = U_aug @ W_aug (double-buffered TDM A-tiles, split-K) ----
__global__ __launch_bounds__(256) void stu_gemm2_kernel(
    const bf16_t* __restrict__ U,  // (L, KW)
    const bf16_t* __restrict__ W,  // (KW, D)
    float* __restrict__ out,       // (L, D)
    int L, int D, int KW, int kChunk) {
  __shared__ alignas(16) bf16_t As[2][TM * SA2];  // TDM-packed, stride 32
  __shared__ alignas(16) bf16_t Bt[2][TN * SBT];

  const int n0 = blockIdx.x * TN;
  const int l0 = blockIdx.y * TM;
  const int k0 = blockIdx.z * kChunk;
  const int k1 = min(KW, k0 + kChunk);

  const int tid  = threadIdx.x;
  const int lane = tid & 31;
  const int wave = tid >> 5;
  const int wm = wave & 3;
  const int wn = wave >> 2;

  auto stageA = [&](int j0, int buf) {
#if HAVE_TDM
    if (tid < 32) {
      tdm_load_2d_bf16(U + (size_t)l0 * KW + j0, lds_offset(&As[buf][0]),
                       (unsigned)KW, (unsigned)L,
                       (unsigned)KSTEP, (unsigned)TM,
                       (unsigned long long)KW);
    }
#else
    for (int e = tid; e < (TM * KSTEP) / 8; e += 256) {
      int i = e >> 2, jg = e & 3;
      uint4 v = *reinterpret_cast<const uint4*>(U + (size_t)(l0 + i) * KW + j0 + jg * 8);
      *reinterpret_cast<uint4*>(&As[buf][i * SA2 + jg * 8]) = v;
    }
#endif
  };
  auto stageB = [&](int j0, int buf) {
    for (int e = tid; e < (KSTEP * TN) / 8; e += 256) {
      int r = e >> 4, cg = e & 15;
      uint4 v = *reinterpret_cast<const uint4*>(W + (size_t)(j0 + r) * D + n0 + cg * 8);
      unsigned w[4] = {v.x, v.y, v.z, v.w};
#pragma unroll
      for (int t = 0; t < 8; ++t) {
        unsigned short h = (unsigned short)((w[t >> 1] >> ((t & 1) * 16)) & 0xFFFFu);
        Bt[buf][(cg * 8 + t) * SBT + r] = __builtin_bit_cast(bf16_t, h);
      }
    }
  };

  v8f acc[2][4];
#pragma unroll
  for (int i = 0; i < 2; ++i)
#pragma unroll
    for (int j = 0; j < 4; ++j) acc[i][j] = (v8f){0,0,0,0,0,0,0,0};

  const int niter = (k1 - k0) / KSTEP;

  stageA(k0, 0);
  stageB(k0, 0);
#if HAVE_TDM
  if (tid < 32) tdm_wait0();
#endif
  __syncthreads();

  for (int it = 0; it < niter; ++it) {
    const int buf = it & 1;
    if (it + 1 < niter) {
      int jn = k0 + (it + 1) * KSTEP;
      stageA(jn, buf ^ 1);   // TDM DMA runs concurrently with staging + WMMA below
      stageB(jn, buf ^ 1);
      __builtin_prefetch(W + (size_t)jn * D + n0, 0, 1);
    }
    v16bf afr[2];
    afr[0] = ldsA16x32(&As[buf][(wm * 32 + 0)  * SA2], SA2, lane);
    afr[1] = ldsA16x32(&As[buf][(wm * 32 + 16) * SA2], SA2, lane);
#pragma unroll
    for (int fn = 0; fn < 4; ++fn) {
      v16bf bfr = ldsB32x16(&Bt[buf][(wn * 64 + fn * 16) * SBT], SBT, lane);
#pragma unroll
      for (int fm = 0; fm < 2; ++fm)
        acc[fm][fn] = __builtin_amdgcn_wmma_f32_16x16x32_bf16(
            false, afr[fm], false, bfr, (short)0, acc[fm][fn], false, false);
    }
    if (it + 1 < niter) {
#if HAVE_TDM
      if (tid < 32) tdm_wait0();   // next A tile landed (hidden behind compute)
#endif
      __syncthreads();
    }
  }

  const int half = lane >> 4, lcol = lane & 15;
#pragma unroll
  for (int fm = 0; fm < 2; ++fm)
#pragma unroll
    for (int fn = 0; fn < 4; ++fn) {
      int r0 = l0 + wm * 32 + fm * 16 + half * 8;
      int c  = n0 + wn * 64 + fn * 16 + lcol;
#pragma unroll
      for (int r = 0; r < 8; ++r)
        atomicAdd(&out[(size_t)(r0 + r) * D + c], acc[fm][fn][r]);
    }
}

// ---------------- host launcher ----------------
extern "C" void kernel_launch(void* const* d_in, const int* in_sizes, int n_in,
                              void* d_out, int out_size, void* d_ws, size_t ws_size,
                              hipStream_t stream) {
  const int L = 2048, D = 768, K = 16, KU = 3;
  const int KW = 2 * K * D + KU * D;  // 26880

  const float* x     = (const float*)d_in[0];
  const float* phi   = (const float*)d_in[1];
  const float* sigma = (const float*)d_in[2];
  const float* Mp    = (const float*)d_in[3];
  const float* Mm    = (const float*)d_in[4];
  const float* Mu    = (const float*)d_in[5];
  float* out = (float*)d_out;

  char* ws = (char*)d_ws;
  size_t off = 0;
  auto alloc = [&](size_t bytes) -> void* {
    void* p = ws + off;
    off = (off + bytes + 255) & ~(size_t)255;
    return p;
  };
  bf16_t* xb   = (bf16_t*)alloc((size_t)L * D * 2);
  bf16_t* phis = (bf16_t*)alloc((size_t)2 * K * L * 2);
  bf16_t* Wb   = (bf16_t*)alloc((size_t)KW * D * 2);
  bf16_t* U    = (bf16_t*)alloc((size_t)L * KW * 2);

  stu_zero_out<<<(L * D + 255) / 256, 256, 0, stream>>>(out, L * D);

  stu_prep_x  <<<(L * D + 255) / 256, 256, 0, stream>>>(x, xb, U, L, D, KW, 2 * K * D);
  stu_prep_phi<<<(2 * K * L + 255) / 256, 256, 0, stream>>>(phi, phis, L, K);
  stu_prep_w  <<<(KW * D + 255) / 256, 256, 0, stream>>>(Mp, Mm, Mu, sigma, Wb, K, D, KW);

  dim3 cg(D / TN, L / TM, 2 * K);
  stu_conv_kernel<<<cg, 256, 0, stream>>>(xb, phis, U, L, D, KW);

  const int SPLITK = 8;
  dim3 gg(D / TN, L / TM, SPLITK);
  stu_gemm2_kernel<<<gg, 256, 0, stream>>>(U, Wb, out, L, D, KW, KW / SPLITK);
}
